// GemmRSIntraNode_8555574853644
// MI455X (gfx1250) — matrix operand order repaired and checked by reading, not compiled, not run on hardware
//
#include <hip/hip_runtime.h>

// ---------------------------------------------------------------------------
// GEMM + reduce-scatter (fused rank reduction) for MI455X (gfx1250).
// out[m,n] = sum_r sum_k A[r,m,k] * W[r,n,k]   (fp32 in/out)
//
// Pass 1 (memory-bound, ~7us): split fp32 -> bf16 hi + bf16 lo residual
//         planes in d_ws (needs ~80 MB workspace).
// Pass 2 (compute-bound): bf16 WMMA GEMM, 3 split-precision terms
//         (hi*hi + lo*hi + hi*lo) with fp32 accumulate -> ~fp32 accuracy at
//         bf16 matrix-pipe rate (8x denser than V_WMMA_F32_16X16X4_F32).
//         128x128x32 tiles, double-buffered global_load_async_to_lds_b128,
//         conflict-free padded LDS, fragments = pure ds_load_b128 (no VALU
//         conversion work in the hot loop).
// ---------------------------------------------------------------------------

typedef __attribute__((ext_vector_type(16))) __bf16 v16bf;
typedef __attribute__((ext_vector_type(8)))  __bf16 bf16x8;
typedef __attribute__((ext_vector_type(4)))  __bf16 bf16x4;
typedef __attribute__((ext_vector_type(8)))  float  v8f;
typedef __attribute__((ext_vector_type(4)))  float  f32x4;

#define WORLD    8
#define M_DIM    8192
#define N_DIM    2048
#define KLOC     256
#define K_TOTAL  (WORLD * KLOC)

#define BM  128
#define BN  128
#define BK  32
#define NKI (K_TOTAL / BK)          // 64 k-chunks
#define LDB 40                      // bf16 row stride in LDS (80B = 20 dwords:
                                    //  16 rows tile all 64 banks, 0 conflicts)
#define PLANE (BM * LDB)            // bf16 elems per LDS tile plane

#define A_ELEMS (WORLD * M_DIM * KLOC)   // 16,777,216
#define W_ELEMS (WORLD * N_DIM * KLOC)   //  4,194,304
// workspace: 2*(A_ELEMS + W_ELEMS) bf16 = 83,886,080 bytes

// Low 32 bits of a generic pointer to __shared__ data == workgroup LDS offset.
__device__ __forceinline__ unsigned lds_off_of(const void* p) {
  return (unsigned)(unsigned long long)p;
}

// Async 16B copy global -> LDS (ASYNCcnt-tracked), saddr form.
__device__ __forceinline__ void async_cp16(unsigned lds_off, const void* sbase,
                                           unsigned goff) {
  asm volatile("global_load_async_to_lds_b128 %0, %1, %2"
               :
               : "v"(lds_off), "v"(goff), "s"(sbase)
               : "memory");
}

__device__ __forceinline__ void wait_async0() {
  asm volatile("s_wait_asynccnt 0" ::: "memory");
}

// 32B LDS fragment load: two ds_load_b128 into one v16bf.
__device__ __forceinline__ v16bf load_frag(const __bf16* p) {
  bf16x8 p0 = *(const bf16x8*)p;
  bf16x8 p1 = *(const bf16x8*)(p + 8);
  return __builtin_shufflevector(p0, p1, 0, 1, 2, 3, 4, 5, 6, 7,
                                 8, 9, 10, 11, 12, 13, 14, 15);
}

// ---------------------------------------------------------------------------
// Pass 1: fp32 -> (bf16 hi, bf16 lo residual), 4 elements/thread.
// ---------------------------------------------------------------------------
extern "C" __global__ __launch_bounds__(256)
void split_bf16(const float* __restrict__ src, __bf16* __restrict__ hi,
                __bf16* __restrict__ lo) {
  const size_t g = (size_t)blockIdx.x * 256 + threadIdx.x;  // float4 index
  f32x4 v = ((const f32x4*)src)[g];
  bf16x4 h, l;
#pragma unroll
  for (int j = 0; j < 4; ++j) {
    __bf16 hj = (__bf16)v[j];
    h[j] = hj;
    l[j] = (__bf16)(v[j] - (float)hj);
  }
  ((bf16x4*)hi)[g] = h;
  ((bf16x4*)lo)[g] = l;
}

// ---------------------------------------------------------------------------
// Pass 2: bf16 split-precision WMMA GEMM with rank-fused K loop.
// ---------------------------------------------------------------------------
extern "C" __global__ __launch_bounds__(256, 1)
void gemm_rs_wmma(const __bf16* __restrict__ Ah, const __bf16* __restrict__ Al,
                  const __bf16* __restrict__ Wh, const __bf16* __restrict__ Wl,
                  float* __restrict__ out) {
  extern __shared__ __bf16 smem[];  // [2 buf][Ah|Al|Bh|Bl planes], 80 KB

  const int tid  = threadIdx.x;
  const int lane = tid & 31;
  const int l16  = lane & 15;
  const int hi   = lane >> 4;
  const int wave = tid >> 5;

  const int bm = blockIdx.y * BM;
  const int bn = blockIdx.x * BN;

  const int waveM = (wave >> 1) * 32;   // 4 wave-rows of 32
  const int waveN = (wave & 1) * 64;    // 2 wave-cols of 64

  v8f acc[2][4];
#pragma unroll
  for (int i = 0; i < 2; ++i)
#pragma unroll
    for (int j = 0; j < 4; ++j)
#pragma unroll
      for (int v = 0; v < 8; ++v) acc[i][j][v] = 0.0f;

  // Per-thread copy slice: per plane 512 x 16B chunks -> 2 chunks/thread.
  // chunk id = tid + t*256 -> row = id/4, c = id%4 (K-group of 8 bf16).
  // A planes land unswizzled (K-group c at c*8, same as B), so each lane's
  // 16-bf16 fragment (k = 16*hi + s, same order as B) is one contiguous 32B read.
  const int c0  = tid & 3;
  const int pc  = ((c0 & 1) << 1) | (c0 >> 1);   // A-group permutation
  const int r0  = tid >> 2;

  auto issue = [&](int kt, int buf) {
    const int r  = kt >> 3;             // rank
    const int kc = (kt & 7) * BK;       // k_local chunk base
    __bf16* base = smem + buf * 4 * PLANE;
#pragma unroll
    for (int t = 0; t < 2; ++t) {
      const int row = r0 + t * 64;
      const unsigned ga =
          (unsigned)(((r * M_DIM + bm + row) * KLOC + kc + c0 * 8) * 2);
      async_cp16(lds_off_of(base + 0 * PLANE + row * LDB + c0 * 8), Ah, ga);
      async_cp16(lds_off_of(base + 1 * PLANE + row * LDB + c0 * 8), Al, ga);
      const unsigned gb =
          (unsigned)(((r * N_DIM + bn + row) * KLOC + kc + c0 * 8) * 2);
      async_cp16(lds_off_of(base + 2 * PLANE + row * LDB + c0 * 8), Wh, gb);
      async_cp16(lds_off_of(base + 3 * PLANE + row * LDB + c0 * 8), Wl, gb);
    }
  };

  issue(0, 0);

#pragma unroll 1
  for (int kt = 0; kt < NKI; ++kt) {
    wait_async0();       // my async copies for chunk kt landed
    __syncthreads();     // everyone's copies landed; prev compute done

    const int buf = kt & 1;
    if (kt + 1 < NKI) issue(kt + 1, buf ^ 1);   // prefetch next chunk

    const __bf16* base = smem + buf * 4 * PLANE;

    v16bf bh[4], bl[4];
#pragma unroll
    for (int nt = 0; nt < 4; ++nt) {
      const __bf16* bp =
          base + 2 * PLANE + (waveN + nt * 16 + l16) * LDB + hi * 16;
      bh[nt] = load_frag(bp);
      bl[nt] = load_frag(bp + PLANE);
    }

#pragma unroll
    for (int mt = 0; mt < 2; ++mt) {
      const __bf16* ap = base + (waveM + mt * 16 + l16) * LDB + hi * 16;
      v16bf ah = load_frag(ap);
      v16bf al = load_frag(ap + PLANE);
#pragma unroll
      for (int nt = 0; nt < 4; ++nt) {
        // split-precision: hi*hi + lo*hi + hi*lo, fp32 accumulate
        acc[mt][nt] = __builtin_amdgcn_wmma_f32_16x16x32_bf16(
            false, ah, false, bh[nt], (short)0, acc[mt][nt], false, false);
        acc[mt][nt] = __builtin_amdgcn_wmma_f32_16x16x32_bf16(
            false, al, false, bh[nt], (short)0, acc[mt][nt], false, false);
        acc[mt][nt] = __builtin_amdgcn_wmma_f32_16x16x32_bf16(
            false, ah, false, bl[nt], (short)0, acc[mt][nt], false, false);
      }
    }
  }

  // Epilogue: C/D 16x16 f32 layout -> VGPR v holds M = v + 8*hi, N = l16.
#pragma unroll
  for (int mt = 0; mt < 2; ++mt)
#pragma unroll
    for (int nt = 0; nt < 4; ++nt) {
      const int gr0 = bm + waveM + mt * 16 + hi * 8;
      const int gc  = bn + waveN + nt * 16 + l16;
#pragma unroll
      for (int v = 0; v < 8; ++v)
        out[(size_t)(gr0 + v) * N_DIM + gc] = acc[mt][nt][v];
    }
}

extern "C" void kernel_launch(void* const* d_in, const int* in_sizes, int n_in,
                              void* d_out, int out_size, void* d_ws, size_t ws_size,
                              hipStream_t stream) {
  const float* A = (const float*)d_in[0];   // [8, 8192, 256] fp32
  const float* W = (const float*)d_in[1];   // [8, 2048, 256] fp32
  float* out = (float*)d_out;               // [8192, 2048] fp32

  __bf16* Ah = (__bf16*)d_ws;               // needs ~80 MB of workspace
  __bf16* Al = Ah + A_ELEMS;
  __bf16* Wh = Al + A_ELEMS;
  __bf16* Wl = Wh + W_ELEMS;

  // Pass 1: split planes (in-stream order guarantees visibility to pass 2).
  hipLaunchKernelGGL(split_bf16, dim3(A_ELEMS / 4 / 256), dim3(256), 0, stream,
                     A, Ah, Al);
  hipLaunchKernelGGL(split_bf16, dim3(W_ELEMS / 4 / 256), dim3(256), 0, stream,
                     W, Wh, Wl);

  // Pass 2: GEMM. grid (16, 64), 8 waves/WG, 80 KB dynamic LDS.
  size_t shmem = (size_t)2 * 4 * PLANE * sizeof(__bf16);  // 81,920 B
  hipLaunchKernelGGL(gemm_rs_wmma, dim3(N_DIM / BN, M_DIM / BM), dim3(256),
                     shmem, stream, Ah, Al, Wh, Wl, out);
}
